// CapsuleLayer_9603546873999
// MI455X (gfx1250) — matrix operand order repaired and checked by reading, not compiled, not run on hardware
//
#include <hip/hip_runtime.h>
#include <hip/hip_bf16.h>
#include <math.h>

typedef __attribute__((ext_vector_type(16))) _Float16 v16h;
typedef __attribute__((ext_vector_type(8)))  float    v8f;

// Problem constants (from reference setup_inputs)
#define BSZ     64      // batch
#define IN_CAPS 512     // input capsules
#define KDIM    128     // in_dim (GEMM K)
#define N_CAPS  16      // output capsules
#define OUT_DIM 32      // out_dim
#define NCOL    512     // N_CAPS*OUT_DIM (GEMM N)

// -------------------------------------------------------------------------
// Kernel 1: u_hat[b,i,j*32+d] = sum_k inputs[b,i,k] * W[i,j,k,d]
// Per input-capsule i: C(64x512) = A(64x128) @ B(128x512), f16 WMMA, f32 acc.
// One wave per (i, 16-row mtile, 32-col j-block): A fragment reused across
// two 16x16x32 WMMAs per K-step; paired B loads consume full 128B W rows.
// u_hat stored as f16 (WMMA-precision data; halves store + routing traffic).
// -------------------------------------------------------------------------
__global__ __launch_bounds__(256)
void uhat_gemm_wmma(const float* __restrict__ inp,
                    const float* __restrict__ W,
                    _Float16* __restrict__ uhat) {
    const int wave = (blockIdx.x * blockDim.x + threadIdx.x) >> 5;
    const int lane = threadIdx.x & 31;

    const int i   = wave >> 6;        // 0..511   (wave / 64)
    const int rem = wave & 63;
    const int mt  = rem >> 4;         // 0..3     (16-row tile of batch)
    const int jb  = rem & 15;         // 0..15    (output capsule = 32-col block)

    const int r16 = lane & 15;
    const int g   = lane >> 4;        // half-wave group

    // A: row m of inputs[:, i, :]
    const int m = mt * 16 + r16;
    const float* Abase = inp + ((size_t)m * IN_CAPS + i) * KDIM;

    // B: W[i, jb] viewed as (K=128) x (D=32); lane owns column d=r16 / d=16+r16
    const float* Bbase = W + (size_t)(i * N_CAPS + jb) * KDIM * OUT_DIM + r16;

    v8f acc0 = {};
    v8f acc1 = {};
#pragma unroll
    for (int kt = 0; kt < 4; ++kt) {
        const int k0 = kt * 32;

        // Prefetch next K-step's B rows (global_prefetch_b8 streaming path)
        if (kt < 3)
            __builtin_prefetch(Bbase + (size_t)(k0 + 32 + g * 16) * OUT_DIM, 0, 1);

        // A fragment: lane (m, g) holds K = k0+g*16+t (t=0..7) and K = k0+g*16+8+t
        v16h a;
        const float* ap0 = Abase + k0 + g * 16;
        const float* ap1 = Abase + k0 + g * 16 + 8;
#pragma unroll
        for (int t = 0; t < 8; ++t) {
            a[t]     = (_Float16)ap0[t];
            a[8 + t] = (_Float16)ap1[t];
        }

        // B fragments: lane (n, g) holds K = k0 + g*16 + t (t=0..15), stride 32
        v16h bf0, bf1;
        const float* bp = Bbase + (size_t)(k0 + g * 16) * OUT_DIM;
#pragma unroll
        for (int t = 0; t < 16; ++t) {
            bf0[t] = (_Float16)bp[(size_t)t * OUT_DIM];        // cols d = 0..15
            bf1[t] = (_Float16)bp[(size_t)t * OUT_DIM + 16];   // cols d = 16..31
        }

        acc0 = __builtin_amdgcn_wmma_f32_16x16x32_f16(
                   false, a, false, bf0, (short)0, acc0, false, false);
        acc1 = __builtin_amdgcn_wmma_f32_16x16x32_f16(
                   false, a, false, bf1, (short)0, acc1, false, false);
    }

    // C layout: VGPR r -> row (r + 8*g) of the 16-row tile, col = r16
#pragma unroll
    for (int r = 0; r < 8; ++r) {
        const int mm = mt * 16 + r + 8 * g;                   // batch row
        const size_t ro = ((size_t)mm * IN_CAPS + i) * NCOL + jb * OUT_DIM;
        uhat[ro + r16]      = (_Float16)acc0[r];
        uhat[ro + 16 + r16] = (_Float16)acc1[r];
    }
}

// -------------------------------------------------------------------------
// Kernel 2: dynamic routing, one workgroup per batch element.
// All routing state (b-coeffs, softmax stats, s, outputs) lives in LDS.
// u_hat (33.5 MB f16) is L2-resident on MI455X (192 MB L2).
// -------------------------------------------------------------------------
__global__ __launch_bounds__(256)
void routing_kernel(const _Float16* __restrict__ uhat,
                    float* __restrict__ out,
                    const int* __restrict__ nroute_p) {
    __shared__ float bco[IN_CAPS * N_CAPS];   // 32 KB routing logits
    __shared__ float maxv[IN_CAPS];
    __shared__ float rden[IN_CAPS];
    __shared__ float sv[NCOL];
    __shared__ float ov[NCOL];
    __shared__ float scal[N_CAPS];

    const int b = blockIdx.x;
    const int t = threadIdx.x;
    const _Float16* U = uhat + (size_t)b * IN_CAPS * NCOL;

    int nroute = *nroute_p;
    if (nroute < 1) nroute = 1;
    if (nroute > 8) nroute = 8;

    for (int e = t; e < IN_CAPS * N_CAPS; e += 256) bco[e] = 0.0f;
    __syncthreads();

    const int jd0 = t, jd1 = t + 256;
    const int j0 = jd0 >> 5, j1 = jd1 >> 5;

    for (int r = 0; r < nroute; ++r) {
        // --- softmax stats over j per input capsule i ---
        for (int i = t; i < IN_CAPS; i += 256) {
            const float* bi = &bco[i * N_CAPS];
            float mx = bi[0];
#pragma unroll
            for (int jj = 1; jj < N_CAPS; ++jj) mx = fmaxf(mx, bi[jj]);
            float s = 0.0f;
#pragma unroll
            for (int jj = 0; jj < N_CAPS; ++jj) s += __expf(bi[jj] - mx);
            maxv[i] = mx;
            rden[i] = 1.0f / s;
        }
        __syncthreads();

        // --- s[j,d] = sum_i c[i,j] * u_hat[i,j,d] (thread owns 2 (j,d) pairs) ---
        {
            float a0 = 0.0f, a1 = 0.0f;
            for (int i = 0; i < IN_CAPS; ++i) {
                const float rm = maxv[i], rd = rden[i];
                const float c0 = __expf(bco[i * N_CAPS + j0] - rm) * rd;
                const float c1 = __expf(bco[i * N_CAPS + j1] - rm) * rd;
                const _Float16* up = U + (size_t)i * NCOL;
                a0 = fmaf(c0, (float)up[jd0], a0);
                a1 = fmaf(c1, (float)up[jd1], a1);
            }
            sv[jd0] = a0;
            sv[jd1] = a1;
        }
        __syncthreads();

        // --- squash scale per output capsule ---
        if (t < N_CAPS) {
            const float* sj = &sv[t * OUT_DIM];
            float ss = 0.0f;
#pragma unroll
            for (int dd = 0; dd < OUT_DIM; ++dd) ss = fmaf(sj[dd], sj[dd], ss);
            scal[t] = ss / (1.0f + ss) * __frsqrt_rn(ss + 1e-7f);
        }
        __syncthreads();
        ov[jd0] = scal[j0] * sv[jd0];
        ov[jd1] = scal[j1] * sv[jd1];
        __syncthreads();

        // --- agreement update b[i,j] += <out[j,:], u_hat[i,j,:]> ---
        if (r < nroute - 1) {
            for (int e = t; e < IN_CAPS * N_CAPS; e += 256) {
                const int ii = e >> 4;
                const int jj = e & 15;
                const _Float16* up = U + (size_t)ii * NCOL + jj * OUT_DIM;
                const float* oj = &ov[jj * OUT_DIM];
                float dot = 0.0f;
#pragma unroll
                for (int dd = 0; dd < OUT_DIM; ++dd)
                    dot = fmaf(oj[dd], (float)up[dd], dot);
                bco[e] += dot;
            }
            __syncthreads();
        }
    }

    out[(size_t)b * NCOL + jd0] = ov[jd0];
    out[(size_t)b * NCOL + jd1] = ov[jd1];
}

// -------------------------------------------------------------------------
extern "C" void kernel_launch(void* const* d_in, const int* in_sizes, int n_in,
                              void* d_out, int out_size, void* d_ws, size_t ws_size,
                              hipStream_t stream) {
    const float* inp = (const float*)d_in[0];   // (64, 512, 128) f32
    const float* W   = (const float*)d_in[1];   // (512, 16, 128, 32) f32
    const int*   nr  = (const int*)d_in[2];     // num_routing scalar

    _Float16* uhat = (_Float16*)d_ws;           // (64, 512, 512) f16 = 33.5 MB
    float* outp = (float*)d_out;                // (64, 16, 32) f32

    // Kernel 1: 512 caps * 4 mtiles * 16 jblocks = 32768 waves; 8 waves/block
    const int waves  = IN_CAPS * 4 * 16;
    const int blocks = waves / 8;
    uhat_gemm_wmma<<<blocks, 256, 0, stream>>>(inp, W, uhat);

    // Kernel 2: one block per batch element
    routing_kernel<<<BSZ, 256, 0, stream>>>(uhat, outp, nr);
}
